// MultiScaleBlock_62947040690720
// MI455X (gfx1250) — matrix operand order repaired and checked by reading, not compiled, not run on hardware
//
#include <hip/hip_runtime.h>
#include <hip/hip_bf16.h>
#include <cstdint>
#include <cstddef>

// ---------------------------------------------------------------------------
// MultiScaleBlock (memory-attention + Mamba) for MI455X / gfx1250 (wave32).
// Big einsums: bf16 WMMA (v_wmma_f32_16x16x32_bf16) GEMMs with double-buffered
// LDS tiles staged by GLOBAL_LOAD_ASYNC_TO_LDS_B128 (ASYNCcnt-tracked DMA).
// 128x128 block tile (8 waves, 32x64 per wave, 8 WMMA / K-step / wave).
// ---------------------------------------------------------------------------

#define Bb 8
#define Lb 2048
#define DM 512
#define DI 1024          // D_INNER
#define BLr (Bb * Lb)    // 16384 rows

typedef __attribute__((ext_vector_type(16))) __bf16 bf16x16;
typedef __attribute__((ext_vector_type(8)))  float  f32x8;

__device__ __forceinline__ unsigned short f2bf_u16(float f) {
  union { float f; unsigned int u; } v; v.f = f;
  unsigned int r = v.u + 0x7FFFu + ((v.u >> 16) & 1u);   // round-nearest-even
  return (unsigned short)(r >> 16);
}

// ---- gfx1250 async memory->LDS DMA (per-lane 16B), ASYNCcnt tracked -------
__device__ __forceinline__ unsigned lds_addr32(const void* p) {
  return (unsigned)(uintptr_t)p;   // low 32 bits of generic ptr = LDS offset
}
__device__ __forceinline__ void async_ld16_to_lds(unsigned lds, const void* gaddr) {
  asm volatile("global_load_async_to_lds_b128 %0, %1, off"
               :: "v"(lds), "v"(gaddr)
               : "memory");
}
__device__ __forceinline__ void wait_asynccnt0() {
  asm volatile("s_wait_asynccnt 0x0" ::: "memory");
}

// ------------------------------ f32 -> bf16 --------------------------------
__global__ __launch_bounds__(256) void k_f32_to_bf16(const float* __restrict__ in,
                                                     unsigned short* __restrict__ out,
                                                     int n) {
  int i = blockIdx.x * 256 + threadIdx.x;
  if (i < n) out[i] = f2bf_u16(in[i]);
}

// ----------------------- WMMA GEMM: C = A(M,K) * W(N,K)^T ------------------
// BNT = 128: block 128x128, wave tile 32x64 (2x4 of 16x16), 8 WMMA / K-step.
// BNT =  64: block 128x64,  wave tile 32x32 (2x2 of 16x16), for small N.
// SOFTPLUS: fused epilogue  C = softplus(acc + bias[n])  (dt_proj GEMM).
#define BM 128
#define BK 32
#define LDA_SH 40   // 32 + 8 pad (80B rows -> conflict-free b128 reads)
#define LDB_SH 40

template<int BNT, bool SOFTPLUS>
__global__ __launch_bounds__(256) void k_gemm_bf16_wmma(
    const unsigned short* __restrict__ A,   // (M,K) bf16 row-major
    const unsigned short* __restrict__ W,   // (N,K) bf16 row-major
    float* __restrict__ C,                  // (M,N) f32 row-major
    const float* __restrict__ bias,         // (N) or nullptr
    int M, int N, int K)
{
  constexpr int NT = BNT / 32;             // n-tiles of 16 per wave
  __shared__ __align__(16) unsigned short sA[2][BM * LDA_SH];
  __shared__ __align__(16) unsigned short sB[2][BNT * LDB_SH];

  const int tid  = threadIdx.x;
  const int lane = tid & 31;
  const int wave = tid >> 5;
  const int wr   = wave & 3;   // 4 waves cover 128 M
  const int wc   = wave >> 2;  // 2 waves cover BNT N
  const int m0   = blockIdx.x * BM;
  const int n0   = blockIdx.y * BNT;

  f32x8 acc[2 * NT] = {};      // [mt*NT + nt]

  const int srow = tid >> 2;        // 0..63, staging row
  const int scol = (tid & 3) * 8;   // ushort offset, 16B chunk

  // per-thread async staging of a 128x32 A tile + BNTx32 W tile
  auto stage = [&](int k0, int b) {
    async_ld16_to_lds(lds_addr32(&sA[b][srow * LDA_SH + scol]),
                      A + (size_t)(m0 + srow) * K + k0 + scol);
    async_ld16_to_lds(lds_addr32(&sA[b][(64 + srow) * LDA_SH + scol]),
                      A + (size_t)(m0 + 64 + srow) * K + k0 + scol);
    async_ld16_to_lds(lds_addr32(&sB[b][srow * LDB_SH + scol]),
                      W + (size_t)(n0 + srow) * K + k0 + scol);
    if constexpr (BNT == 128) {
      async_ld16_to_lds(lds_addr32(&sB[b][(64 + srow) * LDB_SH + scol]),
                        W + (size_t)(n0 + 64 + srow) * K + k0 + scol);
    }
  };

  stage(0, 0);                      // prologue: DMA tile 0

  int buf = 0;
  for (int k0 = 0; k0 < K; k0 += BK, buf ^= 1) {
    // publish tile k (own DMA done -> barrier); also retires all reads of
    // the buffer we are about to overwrite with tile k+1.
    wait_asynccnt0();
    __syncthreads();

    if (k0 + BK < K) stage(k0 + BK, buf ^ 1);   // overlap DMA with compute

    // per-lane fragment loads (contiguous ds_load_b128, ISA A/B layouts)
    const int fr    = lane & 15;
    const int koffA = (lane >> 4) * 8;    // A: lanes16-31 hold K=8..15 / 24..31
    const int koffB = (lane >> 4) * 8;    // B: lanes16-31 hold K=8..15 / 24..31

    bf16x16 a0, a1, bfr[NT];
    {
      const unsigned short* pa0 = &sA[buf][(wr * 32 + fr) * LDA_SH];
      ((uint4*)&a0)[0] = *(const uint4*)(pa0 + koffA);
      ((uint4*)&a0)[1] = *(const uint4*)(pa0 + 16 + koffA);
      const unsigned short* pa1 = &sA[buf][(wr * 32 + 16 + fr) * LDA_SH];
      ((uint4*)&a1)[0] = *(const uint4*)(pa1 + koffA);
      ((uint4*)&a1)[1] = *(const uint4*)(pa1 + 16 + koffA);
      #pragma unroll
      for (int j = 0; j < NT; ++j) {
        const unsigned short* pb =
            &sB[buf][(wc * (BNT / 2) + j * 16 + fr) * LDB_SH + koffB];
        ((uint4*)&bfr[j])[0] = ((const uint4*)pb)[0];
        ((uint4*)&bfr[j])[1] = *(const uint4*)(pb + 16);
      }
    }

    #pragma unroll
    for (int j = 0; j < NT; ++j) {
      acc[j]      = __builtin_amdgcn_wmma_f32_16x16x32_bf16(false, a0, false, bfr[j], (short)0, acc[j],      false, false);
      acc[NT + j] = __builtin_amdgcn_wmma_f32_16x16x32_bf16(false, a1, false, bfr[j], (short)0, acc[NT + j], false, false);
    }
  }

  // C/D layout: lane = (m>=8)*16 + n, vgpr r = m&7
  const int cn    = lane & 15;
  const int mhalf = (lane >> 4) * 8;
  #pragma unroll
  for (int mt = 0; mt < 2; ++mt) {
    #pragma unroll
    for (int j = 0; j < NT; ++j) {
      const int gm = m0 + wr * 32 + mt * 16 + mhalf;
      const int gn = n0 + wc * (BNT / 2) + j * 16 + cn;
      float* cp = C + (size_t)gm * N + gn;
      const f32x8 a = acc[mt * NT + j];
      #pragma unroll
      for (int r = 0; r < 8; ++r) {
        float v = a[r];
        if constexpr (SOFTPLUS) {
          v += bias[gn];
          v = (v > 20.0f) ? v : log1pf(__expf(v));
        }
        cp[(size_t)r * N] = v;
      }
    }
  }
}

// ------------------------ softmax over 512-wide rows -----------------------
__global__ __launch_bounds__(256) void k_softmax512(const float* __restrict__ S,
                                                    unsigned short* __restrict__ P) {
  __shared__ float red[256];
  const int tid = threadIdx.x;
  const size_t row = blockIdx.x;
  const float* sr = S + row * 512;
  float v0 = sr[tid], v1 = sr[tid + 256];
  red[tid] = fmaxf(v0, v1);
  __syncthreads();
  for (int off = 128; off > 0; off >>= 1) {
    if (tid < off) red[tid] = fmaxf(red[tid], red[tid + off]);
    __syncthreads();
  }
  const float m = red[0];
  __syncthreads();
  float e0 = __expf(v0 - m), e1 = __expf(v1 - m);
  red[tid] = e0 + e1;
  __syncthreads();
  for (int off = 128; off > 0; off >>= 1) {
    if (tid < off) red[tid] += red[tid + off];
    __syncthreads();
  }
  const float inv = 1.0f / red[0];
  P[row * 512 + tid]       = f2bf_u16(e0 * inv);
  P[row * 512 + tid + 256] = f2bf_u16(e1 * inv);
}

// -------------------- LayerNorm + exact GELU + residual --------------------
__global__ __launch_bounds__(256) void k_ln_gelu_res(const float* __restrict__ xo,
                                                     const float* __restrict__ x,
                                                     const float* __restrict__ g,
                                                     const float* __restrict__ b,
                                                     unsigned short* __restrict__ out) {
  __shared__ float r1[256], r2[256];
  const int tid = threadIdx.x;
  const size_t row = blockIdx.x;
  const float* p = xo + row * 512;
  float v0 = p[tid], v1 = p[tid + 256];
  r1[tid] = v0 + v1;
  r2[tid] = v0 * v0 + v1 * v1;
  __syncthreads();
  for (int off = 128; off > 0; off >>= 1) {
    if (tid < off) { r1[tid] += r1[tid + off]; r2[tid] += r2[tid + off]; }
    __syncthreads();
  }
  const float mean = r1[0] * (1.0f / 512.0f);
  const float var  = r2[0] * (1.0f / 512.0f) - mean * mean;
  const float inv  = rsqrtf(var + 1e-5f);
  #pragma unroll
  for (int j = 0; j < 2; ++j) {
    int c = tid + j * 256;
    float v = (j == 0) ? v0 : v1;
    float ln = (v - mean) * inv * g[c] + b[c];
    float gl = 0.5f * ln * (1.0f + erff(ln * 0.70710678118654752f));
    out[row * 512 + c] = f2bf_u16(gl + x[row * 512 + c]);
  }
}

// ----------------- depthwise causal conv(4) + bias + SiLU ------------------
__global__ __launch_bounds__(256) void k_conv_silu(const float* __restrict__ xz,
                                                   const float* __restrict__ cw,
                                                   const float* __restrict__ cb,
                                                   float* __restrict__ xact,
                                                   unsigned short* __restrict__ xactbf) {
  int gid = blockIdx.x * 256 + threadIdx.x;
  if (gid >= BLr * DI) return;
  const int d  = gid & (DI - 1);
  const int bl = gid >> 10;
  const int l  = bl & (Lb - 1);
  float acc = cb[d];
  #pragma unroll
  for (int k = 0; k < 4; ++k) {
    int li = l - 3 + k;
    if (li >= 0) acc += cw[d * 4 + k] * xz[(size_t)(bl - 3 + k) * (2 * DI) + d];
  }
  float s = acc / (1.0f + __expf(-acc));   // SiLU
  xact[gid]   = s;
  xactbf[gid] = f2bf_u16(s);
}

// ----------------------- extract dt cols of x_dbl as bf16 ------------------
__global__ __launch_bounds__(256) void k_dt_extract(const float* __restrict__ xdbl,
                                                    unsigned short* __restrict__ dtbf) {
  int gid = blockIdx.x * 256 + threadIdx.x;
  if (gid >= BLr * 32) return;
  int row = gid >> 5, c = gid & 31;
  dtbf[gid] = f2bf_u16(xdbl[(size_t)row * 64 + c]);
}

// -------- selective scan + D residual + SiLU(z) gate, emits bf16 y ---------
// block = 256 channels of one batch; B/C(16 each) broadcast via LDS per step.
__global__ __launch_bounds__(256) void k_scan_fused(const float* __restrict__ dt,
                                                    const float* __restrict__ xact,
                                                    const float* __restrict__ xdbl,
                                                    const float* __restrict__ xz,
                                                    const float* __restrict__ A_log,
                                                    const float* __restrict__ Dp,
                                                    unsigned short* __restrict__ ybf) {
  __shared__ float sB[16], sC[16];
  const int b = blockIdx.x >> 2;                       // DI/256 = 4 blocks per batch
  const int d = ((blockIdx.x & 3) << 8) + threadIdx.x;
  float Arow[16], h[16];
  #pragma unroll
  for (int s = 0; s < 16; ++s) {
    Arow[s] = -__expf(A_log[d * 16 + s]);
    h[s] = 0.0f;
  }
  const float Dd = Dp[d];
  for (int l = 0; l < Lb; ++l) {
    const size_t row = (size_t)b * Lb + l;
    if (threadIdx.x < 32) {
      int s = threadIdx.x & 15;
      if (threadIdx.x < 16) sB[s] = xdbl[row * 64 + 32 + s];
      else                  sC[s] = xdbl[row * 64 + 48 + s];
    }
    __syncthreads();
    const float dt_t = dt[row * DI + d];
    const float x_t  = xact[row * DI + d];
    const float dtx  = dt_t * x_t;
    float y = 0.0f;
    #pragma unroll
    for (int s = 0; s < 16; ++s) {
      float dA = __expf(dt_t * Arow[s]);
      h[s] = h[s] * dA + dtx * sB[s];
      y += h[s] * sC[s];
    }
    const float z = xz[row * (2 * DI) + DI + d];
    const float gate = z / (1.0f + __expf(-z));
    ybf[row * DI + d] = f2bf_u16((y + x_t * Dd) * gate);
    __syncthreads();
  }
}

// ---------------------------------------------------------------------------
extern "C" void kernel_launch(void* const* d_in, const int* in_sizes, int n_in,
                              void* d_out, int out_size, void* d_ws, size_t ws_size,
                              hipStream_t stream) {
  (void)in_sizes; (void)n_in; (void)out_size; (void)ws_size;

  const float* x        = (const float*)d_in[0];
  const float* mk_w     = (const float*)d_in[1];
  const float* mv_w     = (const float*)d_in[2];
  const float* ln_g     = (const float*)d_in[3];
  const float* ln_b     = (const float*)d_in[4];
  const float* in_projw = (const float*)d_in[5];
  const float* conv_w   = (const float*)d_in[6];
  const float* conv_b   = (const float*)d_in[7];
  const float* x_projw  = (const float*)d_in[8];
  const float* dt_projw = (const float*)d_in[9];
  const float* dt_projb = (const float*)d_in[10];
  const float* A_log    = (const float*)d_in[11];
  const float* Dp       = (const float*)d_in[12];
  const float* out_projw= (const float*)d_in[13];
  float* out = (float*)d_out;

  // ---- workspace carve-up (256B aligned blocks) ----
  char* ws = (char*)d_ws;
  size_t o = 0;
  auto take = [&](size_t bytes) { char* p = ws + o; o += (bytes + 255) & ~(size_t)255; return p; };
  unsigned short* xbf      = (unsigned short*)take((size_t)BLr * DM * 2);
  float*          scores   = (float*)take((size_t)BLr * DM * 4);
  unsigned short* probsbf  = (unsigned short*)take((size_t)BLr * DM * 2);
  float*          xout     = (float*)take((size_t)BLr * DM * 4);
  unsigned short* xfusedbf = (unsigned short*)take((size_t)BLr * DM * 2);
  float*          xz       = (float*)take((size_t)BLr * 2 * DI * 4);
  float*          xact     = (float*)take((size_t)BLr * DI * 4);
  unsigned short* xactbf   = (unsigned short*)take((size_t)BLr * DI * 2);
  float*          xdbl     = (float*)take((size_t)BLr * 64 * 4);
  unsigned short* dtbf     = (unsigned short*)take((size_t)BLr * 32 * 2);
  float*          dtf      = (float*)take((size_t)BLr * DI * 4);
  unsigned short* ybf      = (unsigned short*)take((size_t)BLr * DI * 2);
  unsigned short* mkbf     = (unsigned short*)take((size_t)DM * DM * 2);
  unsigned short* mvbf     = (unsigned short*)take((size_t)DM * DM * 2);
  unsigned short* ipbf     = (unsigned short*)take((size_t)2 * DI * DM * 2);
  unsigned short* xpbf     = (unsigned short*)take((size_t)64 * DI * 2);
  unsigned short* dpbf     = (unsigned short*)take((size_t)DI * 32 * 2);
  unsigned short* opbf     = (unsigned short*)take((size_t)DM * DI * 2);

  auto cvt = [&](const float* src, unsigned short* dst, int n) {
    k_f32_to_bf16<<<(n + 255) / 256, 256, 0, stream>>>(src, dst, n);
  };

  // weight + activation conversion to bf16
  cvt(x, xbf, BLr * DM);
  cvt(mk_w, mkbf, DM * DM);
  cvt(mv_w, mvbf, DM * DM);
  cvt(in_projw, ipbf, 2 * DI * DM);
  cvt(x_projw, xpbf, 64 * DI);
  cvt(dt_projw, dpbf, DI * 32);
  cvt(out_projw, opbf, DM * DI);

  // 1) scores = x @ mk_w^T
  k_gemm_bf16_wmma<128, false><<<dim3(BLr / BM, DM / 128), 256, 0, stream>>>(
      xbf, mkbf, scores, nullptr, BLr, DM, DM);
  // 2) softmax rows -> bf16 probs
  k_softmax512<<<BLr, 256, 0, stream>>>(scores, probsbf);
  // 3) x_out = probs @ mv_w^T
  k_gemm_bf16_wmma<128, false><<<dim3(BLr / BM, DM / 128), 256, 0, stream>>>(
      probsbf, mvbf, xout, nullptr, BLr, DM, DM);
  // 4) LN + GELU + residual -> bf16
  k_ln_gelu_res<<<BLr, 256, 0, stream>>>(xout, x, ln_g, ln_b, xfusedbf);
  // 5) xz = xfused @ in_proj^T   (N = 2048)
  k_gemm_bf16_wmma<128, false><<<dim3(BLr / BM, (2 * DI) / 128), 256, 0, stream>>>(
      xfusedbf, ipbf, xz, nullptr, BLr, 2 * DI, DM);
  // 6) causal depthwise conv + SiLU
  k_conv_silu<<<(BLr * DI) / 256, 256, 0, stream>>>(xz, conv_w, conv_b, xact, xactbf);
  // 7) x_dbl = xact @ x_proj^T   (N = 64 -> narrow tile variant)
  k_gemm_bf16_wmma<64, false><<<dim3(BLr / BM, 1), 256, 0, stream>>>(
      xactbf, xpbf, xdbl, nullptr, BLr, 64, DI);
  // 8) dt slice -> bf16
  k_dt_extract<<<(BLr * 32) / 256, 256, 0, stream>>>(xdbl, dtbf);
  // 9) dt_full = softplus(dt @ dt_proj^T + bias)   (K = 32, fused epilogue)
  k_gemm_bf16_wmma<128, true><<<dim3(BLr / BM, DI / 128), 256, 0, stream>>>(
      dtbf, dpbf, dtf, dt_projb, BLr, DI, 32);
  // 10) selective scan + D residual + SiLU(z) gate -> bf16 y
  k_scan_fused<<<(Bb * DI) / 256, 256, 0, stream>>>(dtf, xact, xdbl, xz, A_log, Dp, ybf);
  // 11) out = y @ out_proj^T -> d_out (f32)
  k_gemm_bf16_wmma<128, false><<<dim3(BLr / BM, DM / 128), 256, 0, stream>>>(
      ybf, opbf, out, nullptr, BLr, DM, DI);
}
